// ASSENTGNN_45732811768302
// MI455X (gfx1250) — matrix operand order repaired and checked by reading, not compiled, not run on hardware
//
#include <hip/hip_runtime.h>

// ---------------------------------------------------------------------------
// MI455X (gfx1250) hetero-GNN. Per-edge NNConv weight generation + matvec
// fused into (E x 9216) @ (9216 x 96) GEMM where A = f (x) y outer product
// built in registers (v_pk_mul_bf16); B = w2 flat, bf16, pre-swizzled into
// WMMA B-fragment order. B slabs are double-buffered through LDS with
// GLOBAL_LOAD_ASYNC_TO_LDS_B128 (ASYNCcnt) so all 8 waves of a workgroup
// share one copy; v_wmma_f32_16x16x32_bf16 with f32 accumulate; atomic
// scatter epilogue implements the segment-mean directly.
// ---------------------------------------------------------------------------

typedef __bf16 bf16;
typedef __attribute__((ext_vector_type(16))) __bf16 v16bf;
typedef __attribute__((ext_vector_type(8)))  float  v8f;
typedef __attribute__((ext_vector_type(4)))  int    v4i;

#define HD 96
#define NCHUNK 291                 // 288 w2 chunks + 3 bias(b2) chunks
#define NB_ELEMS (NCHUNK * 6 * 32 * 16)
#define SLAB_ELEMS (3 * 6 * 512)   // one k-slab: 3 hb-chunks x 6 tiles = 9216 bf16
#define SLAB_B128 (SLAB_ELEMS / 8) // 1152 16-byte transfers per slab

static __device__ __forceinline__ void async_ld_b128(const void* g, void* lds)
{
#if __has_builtin(__builtin_amdgcn_global_load_async_to_lds_b128)
    __builtin_amdgcn_global_load_async_to_lds_b128((v4i*)g, (v4i*)lds, 0, 0);
#else
    unsigned off = (unsigned)(__UINTPTR_TYPE__)lds;
    asm volatile("global_load_async_to_lds_b128 %0, %1, off"
                 :: "v"(off), "v"(g) : "memory");
#endif
}

#if __has_builtin(__builtin_amdgcn_s_wait_asynccnt)
#define WAIT_ASYNC(n) __builtin_amdgcn_s_wait_asynccnt(n)
#else
#define WAIT_ASYNC(n) asm volatile("s_wait_asynccnt " #n ::: "memory")
#endif

// ---------------- B-matrix swizzle: w2/b2 (f32) -> bf16 WMMA-B fragments ----
// Layout: Bsw[((kk*6 + t)*32 + lane)*16 + i]
//   lane: n = lane&15 ; element i -> K = (i&7) + (i>>3)*16 + (lane>>4)*8
__global__ __launch_bounds__(256) void k_prep_swizzle(
    const float* __restrict__ w2, const float* __restrict__ b2,
    bf16* __restrict__ Bsw)
{
    int idx = blockIdx.x * blockDim.x + threadIdx.x;
    if (idx >= NB_ELEMS) return;
    int i    = idx & 15;
    int lane = (idx >> 4) & 31;
    int kt   = idx >> 9;          // kk*6 + t
    int t    = kt % 6;
    int kk   = kt / 6;
    int n      = lane & 15;
    int klocal = (i & 7) + ((i >> 3) << 4) + ((lane >> 4) << 3);
    int o      = t * 16 + n;
    float v;
    if (kk < 288) v = w2[(size_t)(kk * 32 + klocal) * HD + o];
    else          v = b2[(size_t)((kk - 288) * 32 + klocal) * HD + o];
    Bsw[idx] = (bf16)v;
}

// ---------------- edge MLP layer 1: f = relu(ea @ w1 + b1) -> bf16 ----------
__global__ __launch_bounds__(HD) void k_edge_mlp1(
    const float* __restrict__ ea, int din,
    const float* __restrict__ w1, const float* __restrict__ b1,
    bf16* __restrict__ f, int E)
{
    int e = blockIdx.x;
    int k = threadIdx.x;
    if (e >= E) return;
    float s = b1[k];
    for (int d = 0; d < din; ++d) s += ea[e * din + d] * w1[d * HD + k];
    f[e * HD + k] = (bf16)fmaxf(s, 0.0f);
}

// ---------------- fused outer-product WMMA GEMM + atomic scatter ------------
// msg[e, :] = (f_e (x) y_e) @ Bsw  then acc[dst[e], :] += msg[e, :]
// Grid must be exactly E/128 blocks (E % 128 == 0): every wave participates
// in async staging and barriers, EXEC is all-ones for WMMA.
__global__ __launch_bounds__(256) void k_wgemm(
    const bf16*  __restrict__ f,      // (E, 96) bf16
    const float* __restrict__ ynode,  // (Ny, 96) f32
    const int*   __restrict__ yidx,   // (E) gather index into ynode
    const int*   __restrict__ didx,   // (E) scatter index into acc
    const bf16*  __restrict__ Bsw,    // swizzled (NCHUNK*6*32*16) bf16
    float*       __restrict__ acc)    // (Nd, 96) f32 accumulator
{
    __shared__ __attribute__((aligned(32))) bf16 Bbuf[2][SLAB_ELEMS]; // 36 KB

    const int tid  = threadIdx.x;
    const int lane = tid & 31;
    const int wave = tid >> 5;
    const int tile = blockIdx.x * 8 + wave;   // 16 edges per wave-tile
    const int m    = lane & 15;
    const int half = lane >> 4;
    const int e    = tile * 16 + m;

    // register cache of this edge's y row, permuted to WMMA A-frag order:
    // yv[hb][i] = y[e, hb*32 + (i%8) + (i/8)*16 + half*8]
    const float* yrow = ynode + (size_t)yidx[e] * HD;
    v16bf yv[3];
#pragma unroll
    for (int hb = 0; hb < 3; ++hb)
#pragma unroll
        for (int i = 0; i < 16; ++i) {
            int kl = (i & 7) + ((i >> 3) << 4) + half * 8;
            yv[hb][i] = (bf16)yrow[hb * 32 + kl];
        }

    const bf16* frow = f + (size_t)e * HD;

    v8f c[6];
#pragma unroll
    for (int t = 0; t < 6; ++t) c[t] = (v8f){0.f,0.f,0.f,0.f,0.f,0.f,0.f,0.f};

    // ---- prologue: async-stage slab 0 into LDS buffer 0 ----
#pragma unroll
    for (int it = 0; it < 5; ++it) {
        int i = tid + it * 256;
        if (i >= SLAB_B128) i -= SLAB_B128;   // wrap -> uniform 5 issues/wave
        async_ld_b128(Bsw + i * 8, &Bbuf[0][i * 8]);
    }

    // K loop: k = 0..95 from f, k == 96 is the b2 bias row (f = 1)
    for (int k = 0; k < 97; ++k) {
        bf16 fk = (k < 96) ? frow[k] : (bf16)1.0f;  // hides under staging

        __syncthreads();   // everyone done reading buf[(k+1)&1] (prev iter)
        if (k + 1 < 97) {
            const bf16* gs = Bsw + (size_t)(k + 1) * SLAB_ELEMS;
            bf16* lb = Bbuf[(k + 1) & 1];
#pragma unroll
            for (int it = 0; it < 5; ++it) {
                int i = tid + it * 256;
                if (i >= SLAB_B128) i -= SLAB_B128;
                async_ld_b128(gs + i * 8, &lb[i * 8]);
            }
            WAIT_ASYNC(5);   // oldest 5 (= slab k) complete; slab k+1 in flight
        } else {
            WAIT_ASYNC(0);   // final slab resident
        }
        __syncthreads();     // slab k visible to all waves

        v16bf fs;
#pragma unroll
        for (int i = 0; i < 16; ++i) fs[i] = fk;

        const bf16* lb = Bbuf[k & 1];
#pragma unroll
        for (int hb = 0; hb < 3; ++hb) {
            v16bf av = yv[hb] * fs;   // v_pk_mul_bf16 x8
#pragma unroll
            for (int t = 0; t < 6; ++t) {
                v16bf bv = *(const v16bf*)&lb[(hb * 6 + t) * 512 + lane * 16];
                c[t] = __builtin_amdgcn_wmma_f32_16x16x32_bf16(
                    false, av, false, bv, (short)0, c[t], false, false);
            }
        }
    }

    // epilogue: C layout lane -> (row = v + 8*half, col = t*16 + (lane&15))
#pragma unroll
    for (int v = 0; v < 8; ++v) {
        int mm = v + half * 8;
        int d  = didx[tile * 16 + mm];
#pragma unroll
        for (int t = 0; t < 6; ++t)
            atomicAdd(&acc[(size_t)d * HD + t * 16 + (lane & 15)], c[t][v]);
    }
}

// ---------------- node encoder: LN(MLP_{din->96->96}(x)) -------------------
__global__ __launch_bounds__(HD) void k_node_encode(
    const float* __restrict__ x, int din,
    const float* __restrict__ w1, const float* __restrict__ b1,
    const float* __restrict__ w2, const float* __restrict__ b2,
    const float* __restrict__ g,  const float* __restrict__ b,
    float* __restrict__ h, int N)
{
    __shared__ float hid[HD];
    __shared__ float out[HD];
    int n = blockIdx.x, j = threadIdx.x;
    if (n >= N) return;
    float s = b1[j];
    for (int d = 0; d < din; ++d) s += x[n * din + d] * w1[d * HD + j];
    hid[j] = fmaxf(s, 0.f);
    __syncthreads();
    float o = b2[j];
    for (int d = 0; d < HD; ++d) o += hid[d] * w2[d * HD + j];
    out[j] = o;
    __syncthreads();
    float m = 0.f, v = 0.f;
    for (int d = 0; d < HD; ++d) m += out[d];
    m *= (1.0f / HD);
    for (int d = 0; d < HD; ++d) { float t = out[d] - m; v += t * t; }
    v *= (1.0f / HD);
    h[n * HD + j] = (o - m) * rsqrtf(v + 1e-5f) * g[j] + b[j];
}

// ---------------- combine: relu( sum_r acc_r/cnt_r + sum_r h@Wr + br ) ------
__global__ __launch_bounds__(HD) void k_combine(
    const float* __restrict__ hold, float* __restrict__ hnew, int N,
    const float* a0, const float* c0, const float* rw0, const float* rb0,
    const float* a1, const float* c1, const float* rw1, const float* rb1,
    const float* a2, const float* c2, const float* rw2, const float* rb2)
{
    __shared__ float row[HD];
    int n = blockIdx.x, o = threadIdx.x;
    if (n >= N) return;
    row[o] = hold[n * HD + o];
    __syncthreads();
    float s = a0[n * HD + o] / fmaxf(c0[n], 1.f) + rb0[o];
    {
        float d0 = 0.f;
        for (int d = 0; d < HD; ++d) d0 += row[d] * rw0[d * HD + o];
        s += d0;
    }
    if (a1) {
        s += a1[n * HD + o] / fmaxf(c1[n], 1.f) + rb1[o];
        float d1 = 0.f;
        for (int d = 0; d < HD; ++d) d1 += row[d] * rw1[d * HD + o];
        s += d1;
    }
    if (a2) {
        s += a2[n * HD + o] / fmaxf(c2[n], 1.f) + rb2[o];
        float d2 = 0.f;
        for (int d = 0; d < HD; ++d) d2 += row[d] * rw2[d * HD + o];
        s += d2;
    }
    hnew[n * HD + o] = fmaxf(s, 0.f);
}

// ---------------- misc helpers ---------------------------------------------
__global__ void k_zero(float* p, int n) {
    int i = blockIdx.x * blockDim.x + threadIdx.x;
    if (i < n) p[i] = 0.f;
}
__global__ void k_count(const int* __restrict__ dst, int E, float* cnt) {
    int e = blockIdx.x * blockDim.x + threadIdx.x;
    if (e < E) atomicAdd(&cnt[dst[e]], 1.0f);
}

// ---------------- heads -----------------------------------------------------
__global__ __launch_bounds__(HD) void k_edge_head(
    const float* __restrict__ h_ap, const float* __restrict__ h_user,
    const float* __restrict__ ea, const int* __restrict__ src,
    const int* __restrict__ dst,
    const float* __restrict__ w1, const float* __restrict__ b1,
    const float* __restrict__ w2, const float* __restrict__ b2,
    float* __restrict__ out, int E)
{
    __shared__ float feat[2 * HD + 6];
    __shared__ float r[HD];
    int e = blockIdx.x, j = threadIdx.x;
    if (e >= E) return;
    feat[j]      = h_ap[src[e] * HD + j];
    feat[HD + j] = h_user[dst[e] * HD + j];
    if (j < 6) feat[2 * HD + j] = ea[e * 6 + j];
    __syncthreads();
    float s = b1[j];
    for (int d = 0; d < 2 * HD + 6; ++d) s += feat[d] * w1[d * HD + j];
    r[j] = fmaxf(s, 0.f) * w2[j];
    __syncthreads();
    if (j == 0) {
        float t = b2[0];
        for (int d = 0; d < HD; ++d) t += r[d];
        out[e] = t;
    }
}

__global__ __launch_bounds__(HD) void k_node_head(
    const float* __restrict__ h,
    const float* __restrict__ w1, const float* __restrict__ b1,
    const float* __restrict__ w2, const float* __restrict__ b2,
    float* __restrict__ out, int N)
{
    __shared__ float r[HD];
    int n = blockIdx.x, j = threadIdx.x;
    if (n >= N) return;
    const float* row = h + (size_t)n * HD;
    float s = b1[j];
    for (int d = 0; d < HD; ++d) s += row[d] * w1[d * HD + j];
    r[j] = fmaxf(s, 0.f) * w2[j];
    __syncthreads();
    if (j == 0) {
        float t = b2[0];
        for (int d = 0; d < HD; ++d) t += r[d];
        out[n] = t;
    }
}

// ===========================================================================
extern "C" void kernel_launch(void* const* d_in, const int* in_sizes, int n_in,
                              void* d_out, int out_size, void* d_ws, size_t ws_size,
                              hipStream_t stream)
{
    const int N_AP = 64, N_USER = 512, N_TGT = 128, E_S = 4096, E_T = 1024;
    auto F = [&](int i) { return (const float*)d_in[i]; };
    auto I = [&](int i) { return (const int*)d_in[i]; };

    // ---- inputs (setup_inputs insertion order) ----
    const float* x_ap  = F(0);
    const float* x_usr = F(1);
    const float* x_tgt = F(2);
    const int* src_s  = I(3);           const int* dst_s  = I(3) + E_S;
    const float* ea_s = F(4);
    const int* src_tx = I(5);           const int* dst_tx = I(5) + E_T;
    const float* ea_tx = F(6);
    const int* src_rx = I(7);           const int* dst_rx = I(7) + E_T;
    const float* ea_rx = F(8);

    // ---- params: jax tree_flatten order (dict keys sorted at each level) ---
    // 9..12  ap_head{b1,b2,w1,w2}   13..16 ap_in{b1,b2,w1,w2}
    // 17..20 edge_head{b1,b2,w1,w2}
    // 21+20*l layers[l]{net_sens{b1,b2,w1,w2}, net_serv{b1,b2,w1,w2},
    //                   rev_rx{b,w}, rev_serves{b,w}, rev_tx{b,w},
    //                   rx{b,w}, serves{b,w}, tx{b,w}}
    // 81,82 ln_ap{b,g} 83,84 ln_tgt{b,g} 85,86 ln_user{b,g}
    // 87..90 target_in  91..94 tgt_head  95..98 user_in
    if (n_in < 99) return;

    // ---- workspace carve ----
    char* wp = (char*)d_ws;
    auto carve = [&](size_t bytes) {
        char* p = wp;
        wp += (bytes + 255) & ~(size_t)255;
        return p;
    };
    float* hap[2]  = {(float*)carve(N_AP * HD * 4),  (float*)carve(N_AP * HD * 4)};
    float* husr[2] = {(float*)carve(N_USER * HD * 4),(float*)carve(N_USER * HD * 4)};
    float* htgt[2] = {(float*)carve(N_TGT * HD * 4), (float*)carve(N_TGT * HD * 4)};
    bf16* f_s  = (bf16*)carve((size_t)E_S * HD * 2);
    bf16* f_tx = (bf16*)carve((size_t)E_T * HD * 2);
    bf16* f_rx = (bf16*)carve((size_t)E_T * HD * 2);
    bf16* Bsw_serv = (bf16*)carve((size_t)NB_ELEMS * 2);
    bf16* Bsw_sens = (bf16*)carve((size_t)NB_ELEMS * 2);
    float* acc_user = (float*)carve(N_USER * HD * 4);
    float* acc_ap0  = (float*)carve(N_AP * HD * 4);
    float* acc_ap1  = (float*)carve(N_AP * HD * 4);
    float* acc_ap2  = (float*)carve(N_AP * HD * 4);
    float* acc_tg0  = (float*)carve(N_TGT * HD * 4);
    float* acc_tg1  = (float*)carve(N_TGT * HD * 4);
    float* cnt_user = (float*)carve(N_USER * 4);
    float* cnt_ap0  = (float*)carve(N_AP * 4);
    float* cnt_ap1  = (float*)carve(N_AP * 4);
    float* cnt_ap2  = (float*)carve(N_AP * 4);
    float* cnt_tg0  = (float*)carve(N_TGT * 4);
    float* cnt_tg1  = (float*)carve(N_TGT * 4);

    auto zero = [&](float* p, int n) {
        k_zero<<<(n + 255) / 256, 256, 0, stream>>>(p, n);
    };

    // ---- encoders ----
    k_node_encode<<<N_AP, HD, 0, stream>>>(x_ap, 7, F(15), F(13), F(16), F(14),
                                           F(82), F(81), hap[0], N_AP);
    k_node_encode<<<N_USER, HD, 0, stream>>>(x_usr, 2, F(97), F(95), F(98), F(96),
                                             F(86), F(85), husr[0], N_USER);
    k_node_encode<<<N_TGT, HD, 0, stream>>>(x_tgt, 2, F(89), F(87), F(90), F(88),
                                            F(84), F(83), htgt[0], N_TGT);

    // ---- per-relation in-degree counts (fixed across layers) ----
    zero(cnt_user, N_USER); zero(cnt_ap0, N_AP); zero(cnt_ap1, N_AP);
    zero(cnt_ap2, N_AP);    zero(cnt_tg0, N_TGT); zero(cnt_tg1, N_TGT);
    k_count<<<(E_S + 255) / 256, 256, 0, stream>>>(dst_s,  E_S, cnt_user);
    k_count<<<(E_S + 255) / 256, 256, 0, stream>>>(src_s,  E_S, cnt_ap0);
    k_count<<<(E_T + 255) / 256, 256, 0, stream>>>(src_tx, E_T, cnt_ap1);
    k_count<<<(E_T + 255) / 256, 256, 0, stream>>>(src_rx, E_T, cnt_ap2);
    k_count<<<(E_T + 255) / 256, 256, 0, stream>>>(dst_tx, E_T, cnt_tg0);
    k_count<<<(E_T + 255) / 256, 256, 0, stream>>>(dst_rx, E_T, cnt_tg1);

    int cur = 0;
    const int swz_blocks = (NB_ELEMS + 255) / 256;
    for (int l = 0; l < 3; ++l) {
        int lb = 21 + 20 * l;
        const float* sens_b1 = F(lb + 0), *sens_b2 = F(lb + 1);
        const float* sens_w1 = F(lb + 2), *sens_w2 = F(lb + 3);
        const float* serv_b1 = F(lb + 4), *serv_b2 = F(lb + 5);
        const float* serv_w1 = F(lb + 6), *serv_w2 = F(lb + 7);
        const float* rrx_b = F(lb + 8),  *rrx_w = F(lb + 9);
        const float* rsv_b = F(lb + 10), *rsv_w = F(lb + 11);
        const float* rtx_b = F(lb + 12), *rtx_w = F(lb + 13);
        const float* rx_b  = F(lb + 14), *rx_w  = F(lb + 15);
        const float* sv_b  = F(lb + 16), *sv_w  = F(lb + 17);
        const float* tx_b  = F(lb + 18), *tx_w  = F(lb + 19);

        // stage bf16 swizzled B matrices (w2 + folded b2 bias rows)
        k_prep_swizzle<<<swz_blocks, 256, 0, stream>>>(serv_w2, serv_b2, Bsw_serv);
        k_prep_swizzle<<<swz_blocks, 256, 0, stream>>>(sens_w2, sens_b2, Bsw_sens);

        // edge MLP hidden activations (bf16)
        k_edge_mlp1<<<E_S, HD, 0, stream>>>(ea_s,  6, serv_w1, serv_b1, f_s,  E_S);
        k_edge_mlp1<<<E_T, HD, 0, stream>>>(ea_tx, 3, sens_w1, sens_b1, f_tx, E_T);
        k_edge_mlp1<<<E_T, HD, 0, stream>>>(ea_rx, 3, sens_w1, sens_b1, f_rx, E_T);

        zero(acc_user, N_USER * HD);
        zero(acc_ap0, N_AP * HD); zero(acc_ap1, N_AP * HD); zero(acc_ap2, N_AP * HD);
        zero(acc_tg0, N_TGT * HD); zero(acc_tg1, N_TGT * HD);

        // six fused outer-product WMMA GEMMs with atomic-mean scatter
        int gS = E_S / 128, gT = E_T / 128;   // exact: all waves full
        k_wgemm<<<gS, 256, 0, stream>>>(f_s,  hap[cur],  src_s,  dst_s,  Bsw_serv, acc_user);
        k_wgemm<<<gS, 256, 0, stream>>>(f_s,  husr[cur], dst_s,  src_s,  Bsw_serv, acc_ap0);
        k_wgemm<<<gT, 256, 0, stream>>>(f_tx, hap[cur],  src_tx, dst_tx, Bsw_sens, acc_tg0);
        k_wgemm<<<gT, 256, 0, stream>>>(f_tx, htgt[cur], dst_tx, src_tx, Bsw_sens, acc_ap1);
        k_wgemm<<<gT, 256, 0, stream>>>(f_rx, hap[cur],  src_rx, dst_rx, Bsw_sens, acc_tg1);
        k_wgemm<<<gT, 256, 0, stream>>>(f_rx, htgt[cur], dst_rx, src_rx, Bsw_sens, acc_ap2);

        // mean + root linears + relu
        int nxt = cur ^ 1;
        k_combine<<<N_USER, HD, 0, stream>>>(husr[cur], husr[nxt], N_USER,
            acc_user, cnt_user, sv_w, sv_b,
            nullptr, nullptr, nullptr, nullptr,
            nullptr, nullptr, nullptr, nullptr);
        k_combine<<<N_AP, HD, 0, stream>>>(hap[cur], hap[nxt], N_AP,
            acc_ap0, cnt_ap0, rsv_w, rsv_b,
            acc_ap1, cnt_ap1, rtx_w, rtx_b,
            acc_ap2, cnt_ap2, rrx_w, rrx_b);
        k_combine<<<N_TGT, HD, 0, stream>>>(htgt[cur], htgt[nxt], N_TGT,
            acc_tg0, cnt_tg0, tx_w, tx_b,
            acc_tg1, cnt_tg1, rx_w, rx_b,
            nullptr, nullptr, nullptr, nullptr);
        cur = nxt;
    }

    // ---- heads: out = [x_logit(4096) | tau_logit(64) | s_logit(128)] ----
    float* out = (float*)d_out;
    k_edge_head<<<E_S, HD, 0, stream>>>(hap[cur], husr[cur], ea_s, src_s, dst_s,
                                        F(19), F(17), F(20), F(18), out, E_S);
    k_node_head<<<N_AP, HD, 0, stream>>>(hap[cur], F(11), F(9), F(12), F(10),
                                         out + E_S, N_AP);
    k_node_head<<<N_TGT, HD, 0, stream>>>(htgt[cur], F(93), F(91), F(94), F(92),
                                          out + E_S + N_AP, N_TGT);
}
